// MemLayer_53008486367424
// MI455X (gfx1250) — matrix operand order repaired and checked by reading, not compiled, not run on hardware
//
#include <hip/hip_runtime.h>

typedef __bf16 v16bf __attribute__((ext_vector_type(16)));
typedef __bf16 v8bf  __attribute__((ext_vector_type(8)));
typedef __bf16 v4bf  __attribute__((ext_vector_type(4)));
typedef float  v8f   __attribute__((ext_vector_type(8)));
typedef unsigned int v4u __attribute__((ext_vector_type(4)));
typedef int   v4i __attribute__((ext_vector_type(4)));
typedef int   v8i __attribute__((ext_vector_type(8)));

#define K_DIM 1024
#define O_DIM 4096
#define B_DIM 4096

static constexpr int BM = 128, BN = 128, BK = 32;
static constexpr int STR = 40;  // bf16 elems per LDS row: 32 + 8 pad (80 B, 16B-aligned)

// ---------------------------------------------------------------------------
// Prep pass: per row (x rows then W rows) compute fp32 sum-of-squares AND
// (optionally) write a bf16 copy of the row into workspace. One block per row,
// 256 threads * float4 = 1024 elems.
// ---------------------------------------------------------------------------
__global__ __launch_bounds__(256) void prep_kernel(
    const float* __restrict__ x, const float* __restrict__ w,
    float* __restrict__ xsq, float* __restrict__ wsq,
    __bf16* __restrict__ xbf, __bf16* __restrict__ wbf, int do_bf)
{
    __shared__ float red[256];
    const int row = blockIdx.x;
    const float* src;
    float* dst;
    __bf16* bdst;
    if (row < B_DIM) {
        src = x + (size_t)row * K_DIM; dst = xsq + row; bdst = xbf + (size_t)row * K_DIM;
    } else {
        const int r = row - B_DIM;
        src = w + (size_t)r * K_DIM; dst = wsq + r; bdst = wbf + (size_t)r * K_DIM;
    }
    const int t = threadIdx.x;
    const float4 v = *(const float4*)(src + t * 4);
    if (do_bf) {
        v4bf b;
        b[0] = (__bf16)v.x; b[1] = (__bf16)v.y; b[2] = (__bf16)v.z; b[3] = (__bf16)v.w;
        *(v4bf*)(bdst + t * 4) = b;
    }
    red[t] = v.x * v.x + v.y * v.y + v.z * v.z + v.w * v.w;
    __syncthreads();
#pragma unroll
    for (int off = 128; off > 0; off >>= 1) {
        if (t < off) red[t] += red[t + off];
        __syncthreads();
    }
    if (t == 0) *dst = red[0];
}

// ---------------------------------------------------------------------------
// Shared compute helpers (fragment loads per ISA layouts + 8 WMMAs + epilogue)
// ---------------------------------------------------------------------------
__device__ __forceinline__ void frag_wmma_step(
    const __bf16* __restrict__ Asb, const __bf16* __restrict__ Bsb,
    int wm, int wn, int h, int l16, v8f acc[4][2])
{
    v16bf a[4], b[2];
#pragma unroll
    for (int i = 0; i < 4; ++i) {
        // A 16x32 bf16: half-wave h: elems 0..7 -> K=8h..8h+7 ; 8..15 -> K=16+8h..+7
        const __bf16* pa = Asb + (wm * 64 + i * 16 + l16) * STR;
        v8bf lo = *(const v8bf*)(pa + 8 * h);
        v8bf hi = *(const v8bf*)(pa + 16 + 8 * h);
        a[i] = __builtin_shufflevector(lo, hi, 0,1,2,3,4,5,6,7,8,9,10,11,12,13,14,15);
    }
#pragma unroll
    for (int j = 0; j < 2; ++j) {
        // B 32x16 bf16: half-wave h: elems 0..7 -> K=8h..8h+7 ; 8..15 -> K=16+8h..+7
        const __bf16* pb = Bsb + (wn * 32 + j * 16 + l16) * STR;
        v8bf lo = *(const v8bf*)(pb + 8 * h);
        v8bf hi = *(const v8bf*)(pb + 16 + 8 * h);
        b[j] = __builtin_shufflevector(lo, hi, 0,1,2,3,4,5,6,7,8,9,10,11,12,13,14,15);
    }
#pragma unroll
    for (int i = 0; i < 4; ++i)
#pragma unroll
        for (int j = 0; j < 2; ++j)
            acc[i][j] = __builtin_amdgcn_wmma_f32_16x16x32_bf16(
                false, a[i], false, b[j], (short)0, acc[i][j], false, false);
}

__device__ __forceinline__ void epilogue_store(
    const v8f acc[4][2], const float* __restrict__ bias,
    const float* __restrict__ xsq, const float* __restrict__ wsq,
    float* __restrict__ out, int gM, int gN, int wm, int wn, int h, int l16)
{
    const float inv = 1.0f / (float)K_DIM;
#pragma unroll
    for (int j = 0; j < 2; ++j) {
        const int o  = gN + wn * 32 + j * 16 + l16;
        const float wq = wsq[o];
        const float bi = bias[o];
#pragma unroll
        for (int i = 0; i < 4; ++i) {
            const int b0 = gM + wm * 64 + i * 16 + 8 * h;  // C/D: VGPR r -> M = r + 8h
#pragma unroll
            for (int r = 0; r < 8; ++r) {
                const int bb = b0 + r;
                const float v = (2.0f * acc[i][j][r] - xsq[bb] - wq) * inv + bi;
                out[(size_t)bb * O_DIM + o] = v;
            }
        }
    }
}

// ---------------------------------------------------------------------------
// TDM tile load: 128 rows x 32 cols of bf16 from a [4096][1024] bf16 tensor
// into LDS with +16B padding per 64B row (-> row stride 80 B = 40 bf16).
// D# per CDNA5 ISA. Issued by one wave; TENSORcnt.
// ---------------------------------------------------------------------------
__device__ __forceinline__ void tdm_load_tile(const __bf16* gsrc, unsigned lds_off)
{
    const unsigned long long ga = (unsigned long long)(uintptr_t)gsrc;
    v4u g0;
    g0[0] = 1u;                                   // count=1, user descriptor
    g0[1] = lds_off;                              // lds_addr (bytes)
    g0[2] = (unsigned)ga;                         // global_addr[31:0]
    g0[3] = (unsigned)((ga >> 32) & 0x1FFFFFFull) // global_addr[56:32]
          | (2u << 30);                           // type=2 ("image")
    v8i g1;
    g1[0] = (int)((1u << 16)    // data_size=1 (2 bytes)
                | (1u << 20)    // pad_enable
                | (3u << 22)    // pad_interval=3 -> every 16 DWORDs (64 B row)
                | (3u << 25));  // pad_amount=3 -> 4 DWORDs (16 B)
    g1[1] = (int)((1024u & 0xFFFFu) << 16);       // tensor_dim0[15:0] (=K)
    g1[2] = (int)(((1024u >> 16) & 0xFFFFu)       // tensor_dim0[31:16]
                | ((4096u & 0xFFFFu) << 16));     // tensor_dim1[15:0] (=rows)
    g1[3] = (int)(((4096u >> 16) & 0xFFFFu)       // tensor_dim1[31:16]
                | (32u << 16));                   // tile_dim0 = 32 (K chunk)
    g1[4] = (int)128u;                            // tile_dim1 = 128 rows, tile_dim2=0
    g1[5] = (int)1024u;                           // tensor_dim0_stride[31:0] (=K)
    g1[6] = 0;                                    // stride hi / dim1_stride lo
    g1[7] = 0;
    const v4i z4 = {0, 0, 0, 0};
#if __has_builtin(__builtin_amdgcn_tensor_load_to_lds)
#if __clang_major__ >= 23
    const v8i z8 = {0, 0, 0, 0, 0, 0, 0, 0};
    __builtin_amdgcn_tensor_load_to_lds(g0, g1, z4, z4, z8, 0);
#else
    __builtin_amdgcn_tensor_load_to_lds(g0, g1, z4, z4, 0);
#endif
#else
    (void)g0; (void)g1; (void)z4;
#endif
}

// ---------------------------------------------------------------------------
// GEMM, bf16 inputs, TDM double-buffered staging.
// ---------------------------------------------------------------------------
__global__ __launch_bounds__(256) void memlayer_gemm_tdm_kernel(
    const __bf16* __restrict__ xbf, const __bf16* __restrict__ wbf,
    const float* __restrict__ bias, const float* __restrict__ xsq,
    const float* __restrict__ wsq, float* __restrict__ out)
{
    __shared__ __attribute__((aligned(16))) __bf16 As[2][BM * STR];
    __shared__ __attribute__((aligned(16))) __bf16 Bs[2][BN * STR];

    const int t    = threadIdx.x;
    const int lane = t & 31;
    const int wave = t >> 5;
    const int wm   = wave >> 2;
    const int wn   = wave & 3;
    const int h    = lane >> 4;
    const int l16  = lane & 15;

    const int gM = blockIdx.y * BM;
    const int gN = blockIdx.x * BN;

    v8f acc[4][2];
#pragma unroll
    for (int i = 0; i < 4; ++i)
#pragma unroll
        for (int j = 0; j < 2; ++j)
#pragma unroll
            for (int e = 0; e < 8; ++e) acc[i][j][e] = 0.f;

    const unsigned ldsA[2] = { (unsigned)(uintptr_t)&As[0][0], (unsigned)(uintptr_t)&As[1][0] };
    const unsigned ldsB[2] = { (unsigned)(uintptr_t)&Bs[0][0], (unsigned)(uintptr_t)&Bs[1][0] };

    // Prologue: DMA first tiles into buffer 0.
    if (wave == 0) {
        tdm_load_tile(xbf + (size_t)gM * K_DIM, ldsA[0]);
        tdm_load_tile(wbf + (size_t)gN * K_DIM, ldsB[0]);
        __builtin_amdgcn_s_wait_tensorcnt(0);
    }
    __syncthreads();

    for (int k0 = 0; k0 < K_DIM; k0 += BK) {
        const int buf = (k0 >> 5) & 1;
        // Prefetch next tiles into the other buffer (safe: everyone finished
        // reading it before the previous end-of-iteration barrier).
        if (wave == 0 && (k0 + BK) < K_DIM) {
            tdm_load_tile(xbf + (size_t)gM * K_DIM + (k0 + BK), ldsA[buf ^ 1]);
            tdm_load_tile(wbf + (size_t)gN * K_DIM + (k0 + BK), ldsB[buf ^ 1]);
        }

        frag_wmma_step(&As[buf][0], &Bs[buf][0], wm, wn, h, l16, acc);

        if (wave == 0) __builtin_amdgcn_s_wait_tensorcnt(0);
        __syncthreads();
    }

    epilogue_store(acc, bias, xsq, wsq, out, gM, gN, wm, wn, h, l16);
}

// ---------------------------------------------------------------------------
// Fallback GEMM (fp32 inputs, convert-in-kernel, single-buffered LDS) — used
// if the workspace is too small for the bf16 copies.
// ---------------------------------------------------------------------------
__global__ __launch_bounds__(256) void memlayer_gemm_kernel(
    const float* __restrict__ x, const float* __restrict__ w,
    const float* __restrict__ bias, const float* __restrict__ xsq,
    const float* __restrict__ wsq, float* __restrict__ out)
{
    __shared__ __attribute__((aligned(16))) __bf16 As[BM * STR];
    __shared__ __attribute__((aligned(16))) __bf16 Bs[BN * STR];

    const int t    = threadIdx.x;
    const int lane = t & 31;
    const int wave = t >> 5;
    const int wm   = wave >> 2;
    const int wn   = wave & 3;
    const int h    = lane >> 4;
    const int l16  = lane & 15;

    const int gM = blockIdx.y * BM;
    const int gN = blockIdx.x * BN;

    v8f acc[4][2];
#pragma unroll
    for (int i = 0; i < 4; ++i)
#pragma unroll
        for (int j = 0; j < 2; ++j)
#pragma unroll
            for (int e = 0; e < 8; ++e) acc[i][j][e] = 0.f;

    const int srow = t >> 3;
    const int scol = t & 7;

    for (int k0 = 0; k0 < K_DIM; k0 += BK) {
#pragma unroll
        for (int p = 0; p < 4; ++p) {
            const int row = p * 32 + srow;
            const float4 xv = *(const float4*)(x + (size_t)(gM + row) * K_DIM + k0 + scol * 4);
            const float4 wv = *(const float4*)(w + (size_t)(gN + row) * K_DIM + k0 + scol * 4);
            v4bf xb, wb;
            xb[0] = (__bf16)xv.x; xb[1] = (__bf16)xv.y; xb[2] = (__bf16)xv.z; xb[3] = (__bf16)xv.w;
            wb[0] = (__bf16)wv.x; wb[1] = (__bf16)wv.y; wb[2] = (__bf16)wv.z; wb[3] = (__bf16)wv.w;
            *(v4bf*)(&As[row * STR + scol * 4]) = xb;
            *(v4bf*)(&Bs[row * STR + scol * 4]) = wb;
        }
        __syncthreads();
        frag_wmma_step(&As[0], &Bs[0], wm, wn, h, l16, acc);
        __syncthreads();
    }

    epilogue_store(acc, bias, xsq, wsq, out, gM, gN, wm, wn, h, l16);
}

extern "C" void kernel_launch(void* const* d_in, const int* in_sizes, int n_in,
                              void* d_out, int out_size, void* d_ws, size_t ws_size,
                              hipStream_t stream) {
    const float* x    = (const float*)d_in[0];   // [4096, 1024]
    const float* w    = (const float*)d_in[1];   // [4096, 1024]
    const float* bias = (const float*)d_in[2];   // [4096]
    float* out = (float*)d_out;                  // [4096, 4096]

    float* xsq = (float*)d_ws;                   // 4096 f32
    float* wsq = xsq + B_DIM;                    // 4096 f32
    const size_t bf_off = 32768;                 // bytes
    __bf16* xbf = (__bf16*)((char*)d_ws + bf_off);
    __bf16* wbf = xbf + (size_t)B_DIM * K_DIM;

    const size_t need = bf_off + 2ull * (size_t)B_DIM * K_DIM * sizeof(__bf16);
    const int use_tdm = (ws_size >= need) ? 1 : 0;

    prep_kernel<<<B_DIM + O_DIM, 256, 0, stream>>>(x, w, xsq, wsq, xbf, wbf, use_tdm);

    dim3 grid(O_DIM / BN, B_DIM / BM);  // (32, 32)
    if (use_tdm) {
        memlayer_gemm_tdm_kernel<<<grid, 256, 0, stream>>>(xbf, wbf, bias, xsq, wsq, out);
    } else {
        memlayer_gemm_kernel<<<grid, 256, 0, stream>>>(x, w, bias, xsq, wsq, out);
    }
}
